// FeatureExtraction_833223655624
// MI455X (gfx1250) — matrix operand order repaired and checked by reading, not compiled, not run on hardware
//
#include <hip/hip_runtime.h>

// Problem constants (from reference): B=8, L=2048, D=256, out channels 2L=4096, taps=5
#define TB   8
#define TL   2048
#define TD   256
#define TL2  4096
#define TAPS 5

typedef __attribute__((ext_vector_type(16))) __bf16 v16bf;
typedef __attribute__((ext_vector_type(8)))  float  v8f;

union FragU { v16bf v; uint4 u[2]; };

__device__ __forceinline__ unsigned short f2bf(float f) {
  unsigned u = __float_as_uint(f);
  unsigned r = u + 0x7FFFu + ((u >> 16) & 1u);   // round-to-nearest-even
  return (unsigned short)(r >> 16);
}

// A fragment (16x32 bf16): lane m = lane&15; kg = lane>>4.
// lanes 0-15 hold K=[kb..kb+15], lanes 16-31 hold K=[kb+16..kb+31] (16 contiguous).
// base is wave-uniform; off is a per-lane 32-bit element offset (row*stride + kk).
__device__ __forceinline__ v16bf load_a(const unsigned short* __restrict__ base,
                                        unsigned off, int lane) {
  unsigned kg8 = ((unsigned)lane >> 4) * 16u;
  FragU f;
  f.u[0] = *reinterpret_cast<const uint4*>(base + off + kg8);
  f.u[1] = *reinterpret_cast<const uint4*>(base + off + kg8 + 8u);
  return f.v;
}

// B fragment (32x16 bf16, stored column-major [N][K]): lane n = lane&15;
// lanes 0-15 hold K=[kb..kb+15], lanes 16-31 hold K=[kb+16..kb+31] (16 contiguous).
__device__ __forceinline__ v16bf load_b(const unsigned short* __restrict__ base,
                                        unsigned off, int lane) {
  unsigned kg16 = ((unsigned)lane >> 4) * 16u;
  const uint4* p = reinterpret_cast<const uint4*>(base + off + kg16);
  FragU f; f.u[0] = p[0]; f.u[1] = p[1];
  return f.v;
}

__device__ __forceinline__ v8f bfwmma(v16bf a, v16bf b, v8f c) {
  return __builtin_amdgcn_wmma_f32_16x16x32_bf16(false, a, false, b, (short)0, c, false, false);
}

// Shared 4x2-subtile (64x32) K-loop: unroll-by-2 -> 16 WMMA + 24 b128 loads per body,
// giving the scheduler independent loads to fill WMMA hazard slots.
template <int KLEN>
__device__ __forceinline__ void gemm_tile_64x32(const unsigned short* __restrict__ abase,
                                                const unsigned* aoff,
                                                const unsigned short* __restrict__ bbase,
                                                const unsigned* boff,
                                                int lane, v8f (&acc)[4][2]) {
  #pragma unroll 2
  for (int kk = 0; kk < KLEN; kk += 32) {
    v16bf a[4], bb[2];
    #pragma unroll
    for (int i = 0; i < 4; ++i) a[i] = load_a(abase, aoff[i] + (unsigned)kk, lane);
    #pragma unroll
    for (int j = 0; j < 2; ++j) bb[j] = load_b(bbase, boff[j] + (unsigned)kk, lane);
    #pragma unroll
    for (int i = 0; i < 4; ++i)
      #pragma unroll
      for (int j = 0; j < 2; ++j) acc[i][j] = bfwmma(a[i], bb[j], acc[i][j]);
  }
}

// ---------------------------------------------------------------- pack / cast
__global__ __launch_bounds__(256) void cvt_f32_bf16(const float* __restrict__ s,
                                                    unsigned short* __restrict__ d, int n) {
  int i = blockIdx.x * 256 + threadIdx.x;
  if (i < n) d[i] = f2bf(s[i]);
}

// w: [O][I][TAPS] fp32  ->  dst: [TAPS][O][I] bf16
__global__ __launch_bounds__(256) void pack_w5(const float* __restrict__ w,
                                               unsigned short* __restrict__ dst,
                                               int O, int I) {
  long i = (long)blockIdx.x * 256 + threadIdx.x;
  long n = (long)O * I * TAPS;
  if (i >= n) return;
  long oi = (long)O * I;
  int  t  = (int)(i / oi);
  long rem = i - (long)t * oi;
  int  o  = (int)(rem / I);
  int  c  = (int)(rem - (long)o * I);
  dst[i] = f2bf(w[((long)o * I + c) * TAPS + t]);
}

// w: [R][C] fp32 -> dst: [C][R] bf16
__global__ __launch_bounds__(256) void pack_T(const float* __restrict__ w,
                                              unsigned short* __restrict__ dst,
                                              int R, int C) {
  int i = blockIdx.x * 256 + threadIdx.x;
  if (i >= R * C) return;
  int c = i / R, r = i - c * R;
  dst[i] = f2bf(w[r * C + c]);
}

// ---------------------------------------------------------------- qkv circular conv (WMMA, 64x32)
__global__ __launch_bounds__(32) void conv_qkv(const unsigned short* __restrict__ xb,
                                               const unsigned short* __restrict__ w3p,
                                               float* __restrict__ q, float* __restrict__ k,
                                               unsigned short* __restrict__ vb) {
  int lane = threadIdx.x;
  int m0 = blockIdx.x * 64;          // row over B*L (tile never straddles a batch)
  int n0 = blockIdx.y * 32;          // dout
  int w  = blockIdx.z;               // 0:q 1:k 2:v
  int b  = m0 / TL;
  int lbase = m0 - b * TL + (lane & 15);
  const unsigned short* wb = w3p + (size_t)w * TAPS * TD * TD;   // [5][dout][din]
  v8f acc[4][2];
  #pragma unroll
  for (int i = 0; i < 4; ++i)
    #pragma unroll
    for (int j = 0; j < 2; ++j) acc[i][j] = (v8f){};
  for (int t = 0; t < TAPS; ++t) {
    unsigned aoff[4], boff[2];
    #pragma unroll
    for (int i = 0; i < 4; ++i) {
      int ls = lbase + i * 16 + t - 2;
      ls += (ls < 0) ? TL : 0;
      ls -= (ls >= TL) ? TL : 0;
      aoff[i] = (unsigned)(b * TL + ls) * TD;
    }
    #pragma unroll
    for (int j = 0; j < 2; ++j)
      boff[j] = (unsigned)(t * TD + n0 + j * 16 + (lane & 15)) * TD;
    gemm_tile_64x32<TD>(xb, aoff, wb, boff, lane, acc);
  }
  int n = lane & 15, mb = (lane >> 4) * 8;
  #pragma unroll
  for (int i = 0; i < 4; ++i)
    #pragma unroll
    for (int j = 0; j < 2; ++j) {
      size_t base = (size_t)(m0 + i * 16 + mb) * TD + n0 + j * 16 + n;
      if (w == 2) {
        #pragma unroll
        for (int r = 0; r < 8; ++r) vb[base + (size_t)r * TD] = f2bf(acc[i][j][r]);
      } else {
        float* dst = (w == 0) ? q : k;
        #pragma unroll
        for (int r = 0; r < 8; ++r) dst[base + (size_t)r * TD] = acc[i][j][r];
      }
    }
}

// ---------------------------------------------------------------- instance norm over L -> bf16
__global__ __launch_bounds__(256) void inorm_bf16(const float* __restrict__ src,
                                                  unsigned short* __restrict__ dst) {
  __shared__ float ssum[256], ssq[256];
  int bd = blockIdx.x;
  int b = bd / TD, d = bd - b * TD;
  const float* p = src + (size_t)b * TL * TD + d;
  float s = 0.f, sq = 0.f;
  for (int i = threadIdx.x; i < TL; i += 256) {
    float v = p[(size_t)i * TD];
    s += v; sq += v * v;
  }
  ssum[threadIdx.x] = s; ssq[threadIdx.x] = sq;
  __syncthreads();
  for (int st = 128; st > 0; st >>= 1) {
    if (threadIdx.x < st) {
      ssum[threadIdx.x] += ssum[threadIdx.x + st];
      ssq[threadIdx.x]  += ssq[threadIdx.x + st];
    }
    __syncthreads();
  }
  float mean = ssum[0] * (1.f / TL);
  float var  = ssq[0] * (1.f / TL) - mean * mean;
  float inv  = rsqrtf(var + 1e-5f);
  unsigned short* o = dst + (size_t)b * TL * TD + d;
  for (int i = threadIdx.x; i < TL; i += 256)
    o[(size_t)i * TD] = f2bf((p[(size_t)i * TD] - mean) * inv);
}

// ---------------------------------------------------------------- scores = q.k^T * scale (WMMA, 64x32)
__global__ __launch_bounds__(32) void scores_gemm(const unsigned short* __restrict__ qb,
                                                  const unsigned short* __restrict__ kbf,
                                                  float* __restrict__ scores) {
  int lane = threadIdx.x;
  int m0 = blockIdx.x * 64;          // l
  int n0 = blockIdx.y * 32;          // s
  int b  = blockIdx.z;
  unsigned aoff[4], boff[2];
  #pragma unroll
  for (int i = 0; i < 4; ++i)
    aoff[i] = (unsigned)(b * TL + m0 + i * 16 + (lane & 15)) * TD;
  #pragma unroll
  for (int j = 0; j < 2; ++j)
    boff[j] = (unsigned)(b * TL + n0 + j * 16 + (lane & 15)) * TD;
  v8f acc[4][2];
  #pragma unroll
  for (int i = 0; i < 4; ++i)
    #pragma unroll
    for (int j = 0; j < 2; ++j) acc[i][j] = (v8f){};
  gemm_tile_64x32<TD>(qb, aoff, kbf, boff, lane, acc);
  const float scale = 0.0625f;       // 1/sqrt(256)
  int n = lane & 15, mb = (lane >> 4) * 8;
  float* out = scores + (size_t)b * TL * TL;
  #pragma unroll
  for (int i = 0; i < 4; ++i)
    #pragma unroll
    for (int j = 0; j < 2; ++j)
      #pragma unroll
      for (int r = 0; r < 8; ++r)
        out[(size_t)(m0 + i * 16 + mb + r) * TL + n0 + j * 16 + n] = acc[i][j][r] * scale;
}

// ---------------------------------------------------------------- softmax over BATCH axis -> bf16
__global__ __launch_bounds__(256) void softmax_batch(const float* __restrict__ scores,
                                                     unsigned short* __restrict__ attnb) {
  size_t idx = (size_t)blockIdx.x * 256 + threadIdx.x;
  const size_t NN = (size_t)TL * TL;
  if (idx >= NN) return;
  float v[TB];
  float mx = -3.4e38f;
  for (int b = 0; b < TB; ++b) { v[b] = scores[(size_t)b * NN + idx]; mx = fmaxf(mx, v[b]); }
  float s = 0.f;
  for (int b = 0; b < TB; ++b) { v[b] = __expf(v[b] - mx); s += v[b]; }
  float inv = 1.f / s;
  for (int b = 0; b < TB; ++b) attnb[(size_t)b * NN + idx] = f2bf(v[b] * inv);
}

// ---------------------------------------------------------------- support^T = (v @ Wg)^T (WMMA, 64x32)
__global__ __launch_bounds__(32) void support_gemm(const unsigned short* __restrict__ vb,
                                                   const unsigned short* __restrict__ wgT,
                                                   unsigned short* __restrict__ supT) {
  int lane = threadIdx.x;
  int m0 = blockIdx.x * 64;          // over B*L
  int n0 = blockIdx.y * 32;          // f
  int b  = m0 / TL;
  unsigned aoff[4], boff[2];
  #pragma unroll
  for (int i = 0; i < 4; ++i)
    aoff[i] = (unsigned)(m0 + i * 16 + (lane & 15)) * TD;
  #pragma unroll
  for (int j = 0; j < 2; ++j)
    boff[j] = (unsigned)(n0 + j * 16 + (lane & 15)) * TD;
  v8f acc[4][2];
  #pragma unroll
  for (int i = 0; i < 4; ++i)
    #pragma unroll
    for (int j = 0; j < 2; ++j) acc[i][j] = (v8f){};
  gemm_tile_64x32<TD>(vb, aoff, wgT, boff, lane, acc);
  int n = lane & 15, mb = (lane >> 4) * 8;
  int lloc = m0 - b * TL;
  #pragma unroll
  for (int i = 0; i < 4; ++i)
    #pragma unroll
    for (int j = 0; j < 2; ++j)
      #pragma unroll
      for (int r = 0; r < 8; ++r)
        supT[((size_t)b * TD + n0 + j * 16 + n) * TL + lloc + i * 16 + mb + r] = f2bf(acc[i][j][r]);
}

// -------------------------------------------- gcn = attn@support; h^T = bf16(relu(gcn+bg)+x)  (64x32)
__global__ __launch_bounds__(32) void gcn_gemm(const unsigned short* __restrict__ attnb,
                                               const unsigned short* __restrict__ supT,
                                               const float* __restrict__ bg,
                                               const float* __restrict__ x,
                                               unsigned short* __restrict__ hT) {
  int lane = threadIdx.x;
  int m0 = blockIdx.x * 64;          // s
  int n0 = blockIdx.y * 32;          // e
  int b  = blockIdx.z;
  const unsigned short* abase = attnb + (size_t)b * TL * TL;
  unsigned aoff[4], boff[2];
  #pragma unroll
  for (int i = 0; i < 4; ++i)
    aoff[i] = (unsigned)(m0 + i * 16 + (lane & 15)) * TL;
  #pragma unroll
  for (int j = 0; j < 2; ++j)
    boff[j] = (unsigned)(b * TD + n0 + j * 16 + (lane & 15)) * TL;
  v8f acc[4][2];
  #pragma unroll
  for (int i = 0; i < 4; ++i)
    #pragma unroll
    for (int j = 0; j < 2; ++j) acc[i][j] = (v8f){};
  gemm_tile_64x32<TL>(abase, aoff, supT, boff, lane, acc);
  int n = lane & 15, mb = (lane >> 4) * 8;
  #pragma unroll
  for (int j = 0; j < 2; ++j) {
    int e = n0 + j * 16 + n;
    float bias = bg[e];
    #pragma unroll
    for (int i = 0; i < 4; ++i)
      #pragma unroll
      for (int r = 0; r < 8; ++r) {
        int s = m0 + i * 16 + mb + r;
        float g = fmaxf(acc[i][j][r] + bias, 0.f) + x[((size_t)b * TL + s) * TD + e];
        hT[((size_t)b * TD + e) * TL + s] = f2bf(g);   // h^T: [b][t=e][c=s]
      }
  }
}

// ---------------------------------------- out = circular conv over D with C=L (5 GEMMs/tap, 64x32)
__global__ __launch_bounds__(32) void outconv_gemm(const unsigned short* __restrict__ woutp,
                                                   const unsigned short* __restrict__ hT,
                                                   float* __restrict__ out) {
  int lane = threadIdx.x;
  int m0 = blockIdx.x * 64;          // o in [0, 2L)
  int n0 = blockIdx.y * 32;          // t in [0, D)
  int b  = blockIdx.z;
  v8f acc[4][2];
  #pragma unroll
  for (int i = 0; i < 4; ++i)
    #pragma unroll
    for (int j = 0; j < 2; ++j) acc[i][j] = (v8f){};
  for (int tap = 0; tap < TAPS; ++tap) {
    unsigned aoff[4], boff[2];
    #pragma unroll
    for (int i = 0; i < 4; ++i)
      aoff[i] = (unsigned)(tap * TL2 + m0 + i * 16 + (lane & 15)) * TL;  // [5][o][c]
    #pragma unroll
    for (int j = 0; j < 2; ++j) {
      int tc = n0 + j * 16 + (lane & 15) + tap - 2;
      tc += (tc < 0) ? TD : 0;
      tc -= (tc >= TD) ? TD : 0;
      boff[j] = (unsigned)(b * TD + tc) * TL;                            // h^T[b][t'][c]
    }
    gemm_tile_64x32<TL>(woutp, aoff, hT, boff, lane, acc);
  }
  int n = lane & 15, mb = (lane >> 4) * 8;
  #pragma unroll
  for (int i = 0; i < 4; ++i)
    #pragma unroll
    for (int j = 0; j < 2; ++j)
      #pragma unroll
      for (int r = 0; r < 8; ++r)
        out[((size_t)b * TL2 + m0 + i * 16 + mb + r) * TD + n0 + j * 16 + n] = acc[i][j][r];
}

// ---------------------------------------------------------------- launch
extern "C" void kernel_launch(void* const* d_in, const int* in_sizes, int n_in,
                              void* d_out, int out_size, void* d_ws, size_t ws_size,
                              hipStream_t stream) {
  const float* x    = (const float*)d_in[0];   // [B, L, D]
  const float* Wq   = (const float*)d_in[1];   // [D, D, 5]
  const float* Wk   = (const float*)d_in[2];
  const float* Wv   = (const float*)d_in[3];
  const float* Wg   = (const float*)d_in[4];   // [D, D]
  const float* bg   = (const float*)d_in[5];   // [D]
  const float* Wout = (const float*)d_in[6];   // [2L, L, 5]
  float* out = (float*)d_out;                  // [B, 2L, D]

  // workspace carve-up (~371 MB total)
  char* ws = (char*)d_ws;
  size_t off = 0;
  auto carve = [&](size_t bytes) { char* p = ws + off; off += (bytes + 255) & ~(size_t)255; return p; };
  const size_t NXD = (size_t)TB * TL * TD;            // 4,194,304
  unsigned short* xb    = (unsigned short*)carve(NXD * 2);
  unsigned short* w3p   = (unsigned short*)carve((size_t)3 * TAPS * TD * TD * 2);
  unsigned short* wgT   = (unsigned short*)carve((size_t)TD * TD * 2);
  unsigned short* woutp = (unsigned short*)carve((size_t)TAPS * TL2 * TL * 2);
  float*          q     = (float*)carve(NXD * 4);
  float*          k     = (float*)carve(NXD * 4);
  unsigned short* qb    = (unsigned short*)carve(NXD * 2);
  unsigned short* kb    = (unsigned short*)carve(NXD * 2);
  unsigned short* vb    = (unsigned short*)carve(NXD * 2);
  float*          sc    = (float*)carve((size_t)TB * TL * TL * 4);
  unsigned short* attnb = (unsigned short*)carve((size_t)TB * TL * TL * 2);
  unsigned short* supT  = (unsigned short*)carve(NXD * 2);
  unsigned short* hT    = (unsigned short*)carve(NXD * 2);
  (void)ws_size; (void)in_sizes; (void)n_in; (void)out_size;

  // 1) casts / packs
  cvt_f32_bf16<<<(int)((NXD + 255) / 256), 256, 0, stream>>>(x, xb, (int)NXD);
  {
    int nw = TAPS * TD * TD, g = (nw + 255) / 256;
    pack_w5<<<g, 256, 0, stream>>>(Wq, w3p + 0 * (size_t)nw, TD, TD);
    pack_w5<<<g, 256, 0, stream>>>(Wk, w3p + 1 * (size_t)nw, TD, TD);
    pack_w5<<<g, 256, 0, stream>>>(Wv, w3p + 2 * (size_t)nw, TD, TD);
  }
  pack_T<<<(TD * TD + 255) / 256, 256, 0, stream>>>(Wg, wgT, TD, TD);
  {
    long nw = (long)TAPS * TL2 * TL;
    pack_w5<<<(int)((nw + 255) / 256), 256, 0, stream>>>(Wout, woutp, TL2, TL);
  }

  // 2) q/k/v circular conv (WMMA bf16, 64x32 tiles)
  conv_qkv<<<dim3(TB * TL / 64, TD / 32, 3), 32, 0, stream>>>(xb, w3p, q, k, vb);

  // 3) instance-norm over L -> bf16
  inorm_bf16<<<TB * TD, 256, 0, stream>>>(q, qb);
  inorm_bf16<<<TB * TD, 256, 0, stream>>>(k, kb);

  // 4) scores (WMMA) and batch-axis softmax -> bf16 attn
  scores_gemm<<<dim3(TL / 64, TL / 32, TB), 32, 0, stream>>>(qb, kb, sc);
  softmax_batch<<<(int)(((size_t)TL * TL + 255) / 256), 256, 0, stream>>>(sc, attnb);

  // 5) support = v @ Wg (WMMA, transposed bf16 store)
  support_gemm<<<dim3(TB * TL / 64, TD / 32), 32, 0, stream>>>(vb, wgT, supT);

  // 6) gcn = attn @ support + bias, relu, residual -> h^T bf16
  gcn_gemm<<<dim3(TL / 64, TD / 32, TB), 32, 0, stream>>>(attnb, supT, bg, x, hT);

  // 7) final circular conv over D with C=L (WMMA) -> out fp32
  outconv_gemm<<<dim3(TL2 / 64, TD / 32, TB), 32, 0, stream>>>(woutp, hT, out);
}
